// CleanLC0Model_80101140070424
// MI455X (gfx1250) — matrix operand order repaired and checked by reading, not compiled, not run on hardware
//
#include <hip/hip_runtime.h>

// ----------------------------------------------------------------------------
// LC0 transformer forward on MI455X (gfx1250, wave32, WMMA).
//
// Roofline: ~4.4 TFLOP of dense GEMM per forward vs ~1-2 GB of traffic =>
// matrix-core bound. Design: bf16 end-to-end (weights pre-transposed K-major
// bf16 once per call via an LDS-tiled transpose), v_wmma_f32_16x16x32_bf16
// with fp32 accumulation, 128x128 block tiles (64x64 per wave, 16 WMMA per
// K-step), double-buffered LDS fed by GLOBAL_LOAD_ASYNC_TO_LDS_B128
// (ASYNCcnt), and TDM tensor_load_to_lds for attention Q/V/SW tiles
// (TENSORcnt).
//
// WMMA fragment layouts (CDNA5 ISA 7.12.2):
//   A (16x32 bf16): lane = M row; VGPR/half = K (lanes 0-15: K{0..7,16..23},
//     lanes 16-31: K{8..15,24..31}) -> two b128 LDS reads from [m][k] tiles.
//   B (32x16 bf16): lane = N col; VGPR/half = K (same split as A) -> strided reads from [k][n].
//   C/D (16x16 f32): lane = N; VGPR r = M row r (lanes 0-15) / r+8 (16-31).
// ----------------------------------------------------------------------------

typedef __bf16 bf16;
typedef __attribute__((ext_vector_type(16))) __bf16 bhalf16;
typedef __attribute__((ext_vector_type(8)))  float  floatx8;
typedef unsigned int u32x4 __attribute__((ext_vector_type(4)));
typedef int i32x4 __attribute__((ext_vector_type(4)));
typedef int i32x8 __attribute__((ext_vector_type(8)));
typedef int v4i_t __attribute__((vector_size(16)));
typedef void as1_void __attribute__((address_space(1)));
typedef void as3_void __attribute__((address_space(3)));
typedef __attribute__((address_space(1))) v4i_t as1_v4i;
typedef __attribute__((address_space(3))) v4i_t as3_v4i;

#define NB  512
#define NS  64
#define ND  768
#define NH  24
#define NDK 32
#define NFF 1024
#define NL  15
#define NTOK (NB*NS)

// ---------------------------- CDNA5 feature probes ---------------------------
#if __has_builtin(__builtin_amdgcn_global_load_async_to_lds_b128)
#define USE_ASYNC_COPY 1
#else
#define USE_ASYNC_COPY 0
#endif

#if __has_builtin(__builtin_amdgcn_tensor_load_to_lds) && \
    __has_builtin(__builtin_amdgcn_s_wait_tensorcnt)
#define HAVE_TDM 1
#else
#define HAVE_TDM 0
#endif

__device__ inline void async_cp16(const bf16* g, bf16* l) {
#if USE_ASYNC_COPY
  __builtin_amdgcn_global_load_async_to_lds_b128(
      (as1_v4i*)(as1_void*)(void*)const_cast<bf16*>(g),
      (as3_v4i*)(as3_void*)(void*)l, 0, 0);
#else
  *(uint4*)l = *(const uint4*)g;  // synchronous fallback through VGPRs
#endif
}

__device__ inline void wait_async_all() {
#if USE_ASYNC_COPY
#if __has_builtin(__builtin_amdgcn_s_wait_asynccnt)
  __builtin_amdgcn_s_wait_asynccnt(0);
#else
  asm volatile("s_wait_asynccnt 0x0" ::: "memory");
#endif
#endif
}

#if HAVE_TDM
// Build a 2D-tile D# (ISA 8.3/8.4) and issue tensor_load_to_lds.
// tileW elems per row (bf16), tileH rows, rowStride elems, optional LDS pad of
// 4 dwords after every 16 dwords (-> 80B padded rows matching [..][40] tiles).
__device__ inline void tdm_load_2d(unsigned ldsByteOff, const void* gaddr,
                                   unsigned tileW, unsigned tileH,
                                   unsigned rowStride, int padded) {
  unsigned long long ga = (unsigned long long)gaddr;
  u32x4 g0 = {};
  g0[0] = 1u;                                     // count=1 (valid descriptor)
  g0[1] = ldsByteOff;                             // lds_addr
  g0[2] = (unsigned)ga;                           // global_addr[31:0]
  g0[3] = (unsigned)((ga >> 32) & 0x01FFFFFFull)  // global_addr[56:32]
          | (2u << 30);                           // type=2 (image)
  i32x8 g1 = {};
  unsigned w0 = (1u << 16);                       // data_size = 2 bytes
  if (padded) w0 |= (1u << 20) | (3u << 22) | (3u << 25);  // pad 4dw per 16dw
  g1[0] = (int)w0;
  g1[1] = (int)((rowStride & 0xFFFFu) << 16);     // tensor_dim0[15:0]
  g1[2] = (int)((rowStride >> 16) | ((tileH & 0xFFFFu) << 16)); // dim0 hi, dim1 lo
  g1[3] = (int)((tileW & 0xFFFFu) << 16);         // dim1 hi (0) | tile_dim0
  g1[4] = (int)(tileH & 0xFFFFu);                 // tile_dim1 (tile_dim2 = 0)
  g1[5] = (int)rowStride;                         // tensor_dim0_stride[31:0]
  i32x4 z4 = {};
#if __clang_major__ >= 23
  i32x8 z8 = {};
  __builtin_amdgcn_tensor_load_to_lds(g0, g1, z4, z4, z8, 0);
#else
  __builtin_amdgcn_tensor_load_to_lds(g0, g1, z4, z4, 0);
#endif
}
#endif

// ------------------------------- WMMA helpers --------------------------------
union FragU { bhalf16 v; uint4 q[2]; };

__device__ inline bhalf16 load_a_frag(const bf16* rowBase, int halfK) {
  FragU u;
  const uint4* p = (const uint4*)rowBase;
  u.q[0] = p[halfK];
  u.q[1] = p[2 + halfK];
  return u.v;
}
__device__ inline bhalf16 load_b_frag(const bf16* kRowAtN, int ld) {
  bhalf16 v;
#pragma unroll
  for (int s = 0; s < 16; ++s) v[s] = kRowAtN[(s < 8 ? s : s + 8) * ld];
  return v;
}
__device__ inline floatx8 wmma_bf16(bhalf16 a, bhalf16 b, floatx8 c) {
  return __builtin_amdgcn_wmma_f32_16x16x32_bf16(false, a, false, b,
                                                 (short)0, c, false, false);
}

__device__ inline float mish_f(float x) {
  float sp = __logf(1.0f + __expf(x));
  return x * tanhf(sp);
}
__device__ inline float silu_f(float x) { return x * (1.0f / (1.0f + __expf(-x))); }

enum { ACT_NONE = 0, ACT_MISH = 1, ACT_SILU = 2, ACT_SQRELU = 3 };

template <int ACT>
__device__ inline float act_fn(float x) {
  if constexpr (ACT == ACT_MISH)        return mish_f(x);
  else if constexpr (ACT == ACT_SILU)   return silu_f(x);
  else if constexpr (ACT == ACT_SQRELU) { float r = fmaxf(x, 0.0f); return r * r; }
  else return x;
}

// ----------------------------------------------------------------------------
// Weight prep: fp32 [L,N,K] row-major -> bf16 [L,K,Np] K-major, zero padded.
// LDS-tiled 32x32 transpose: coalesced reads (along k) and writes (along n),
// no integer division. Grid (Np/32, K/32, L), block (32,8).
// ----------------------------------------------------------------------------
__global__ __launch_bounds__(256)
void transpose_tile_kernel(const float* __restrict__ in, bf16* __restrict__ out,
                           int N, int K, int Np) {
  __shared__ float tile[32][33];
  const int l = blockIdx.z;
  const int n0 = blockIdx.x * 32, k0 = blockIdx.y * 32;
  const int tx = threadIdx.x, ty = threadIdx.y;
#pragma unroll
  for (int j = 0; j < 4; ++j) {
    int n = n0 + ty + j * 8;
    float v = (n < N) ? in[((size_t)l * N + n) * K + k0 + tx] : 0.0f;
    tile[ty + j * 8][tx] = v;   // tile[n_local][k_local]
  }
  __syncthreads();
#pragma unroll
  for (int j = 0; j < 4; ++j) {
    int k = k0 + ty + j * 8;
    out[((size_t)l * K + k) * Np + n0 + tx] = (bf16)tile[tx][ty + j * 8];
  }
}

// ----------------------------------------------------------------------------
// GEMM: out[M,N] = act(A[M,K] @ Wt[K,Np](cols 0..N) + bias[N])
// A bf16 row-major, Wt bf16 K-major (pre-transposed, padded to Np).
// 128x128 block, BK=32, 4 waves x (64x64), double-buffered async LDS staging.
// ----------------------------------------------------------------------------
template <int ACT, typename OutT>
__global__ __launch_bounds__(128)
void gemm_v2(const bf16* __restrict__ A, const bf16* __restrict__ Wt,
             const float* __restrict__ bias, OutT* __restrict__ out,
             int M, int N, int Np, int K) {
  __shared__ bf16 As[2][128][40];   // [m][k], 80B padded rows
  __shared__ bf16 Bs[2][32][136];   // [k][n], 272B padded rows

  const int tid = threadIdx.x;
  const int wave = tid >> 5, lane = tid & 31;
  const int l16 = lane & 15, hk = lane >> 4;
  const int m0 = blockIdx.y * 128, n0 = blockIdx.x * 128;
  const int wm = (wave >> 1) * 64, wn = (wave & 1) * 64;

  floatx8 acc[4][4] = {};

  auto stage = [&](int buf, int k0) {
#pragma unroll
    for (int i = 0; i < 4; ++i) {           // A: 128 rows x 64B
      int v = tid + i * 128;
      int r = v >> 2, c = (v & 3) * 8;
      async_cp16(A + (size_t)(m0 + r) * K + k0 + c, &As[buf][r][c]);
    }
#pragma unroll
    for (int i = 0; i < 4; ++i) {           // B: 32 k-rows x 256B
      int v = tid + i * 128;
      int kr = v >> 4, c = (v & 15) * 8;
      async_cp16(Wt + (size_t)(k0 + kr) * Np + n0 + c, &Bs[buf][kr][c]);
    }
  };

  int buf = 0;
  stage(0, 0);
  for (int k0 = 0; k0 < K; k0 += 32) {
    wait_async_all();
    __syncthreads();                         // tile[buf] visible to all waves
    if (k0 + 32 < K) stage(buf ^ 1, k0 + 32);
    bhalf16 bfr[4];
#pragma unroll
    for (int fn = 0; fn < 4; ++fn)
      bfr[fn] = load_b_frag(&Bs[buf][hk * 8][wn + fn * 16 + l16], 136);
#pragma unroll
    for (int fm = 0; fm < 4; ++fm) {
      bhalf16 af = load_a_frag(&As[buf][wm + fm * 16 + l16][0], hk);
#pragma unroll
      for (int fn = 0; fn < 4; ++fn)
        acc[fm][fn] = wmma_bf16(af, bfr[fn], acc[fm][fn]);
    }
    __syncthreads();                         // done reading tile[buf]
    buf ^= 1;
  }

#pragma unroll
  for (int fm = 0; fm < 4; ++fm) {
#pragma unroll
    for (int fn = 0; fn < 4; ++fn) {
      int col = n0 + wn + fn * 16 + l16;
      if (col < N) {
        float bv = bias ? bias[col] : 0.0f;
#pragma unroll
        for (int r = 0; r < 8; ++r) {
          int row = m0 + wm + fm * 16 + r + hk * 8;
          if (row < M) {
            float val = act_fn<ACT>(acc[fm][fn][r] + bv);
            out[(size_t)row * N + col] = (OutT)val;
          }
        }
      }
    }
  }
}

// ----------------------------------------------------------------------------
// Embedding (K=176, VALU): x = mish(concat(board,pos) @ emb_w^T + b)*gmul+gadd
// ----------------------------------------------------------------------------
__global__ __launch_bounds__(256)
void embed_kernel(const float* __restrict__ board, const float* __restrict__ pos,
                  const float* __restrict__ emb_w, const float* __restrict__ emb_b,
                  const float* __restrict__ gmul, const float* __restrict__ gadd,
                  bf16* __restrict__ X) {
  __shared__ float feat[176];
  const int row = blockIdx.x;
  const int b = row >> 6, s = row & 63;
  if (threadIdx.x < 176) {
    float v;
    if (threadIdx.x < 112)
      v = board[((size_t)b * 112 + threadIdx.x) * 64 + s];
    else
      v = pos[(size_t)s * 64 + (threadIdx.x - 112)];
    feat[threadIdx.x] = v;
  }
  __syncthreads();
  for (int d = threadIdx.x; d < ND; d += 256) {
    const float* wr = emb_w + (size_t)d * 176;
    float acc = emb_b[d];
#pragma unroll 4
    for (int k = 0; k < 176; ++k) acc = fmaf(feat[k], wr[k], acc);
    float m = mish_f(acc);
    X[(size_t)row * ND + d] =
        (bf16)(m * gmul[(size_t)s * ND + d] + gadd[(size_t)s * ND + d]);
  }
}

// ----------------------------------------------------------------------------
// LayerNorm (bf16 in/out, fp32 math): out = ln(a + alpha*res)*g + beta
// ----------------------------------------------------------------------------
__global__ __launch_bounds__(256)
void ln_kernel(const bf16* __restrict__ a, const bf16* __restrict__ res,
               const float* __restrict__ alpha, const float* __restrict__ g,
               const float* __restrict__ beta, bf16* __restrict__ out, int N) {
  const int row = blockIdx.x;
  const bf16* pa = a + (size_t)row * N;
  const bf16* pr = res ? res + (size_t)row * N : nullptr;
  const float al = alpha ? alpha[0] : 0.0f;

  float s = 0.f, s2 = 0.f;
  for (int i = threadIdx.x; i < N; i += 256) {
    float v = (float)pa[i] + (pr ? al * (float)pr[i] : 0.0f);
    s += v; s2 += v * v;
  }
#pragma unroll
  for (int o = 16; o > 0; o >>= 1) { s += __shfl_down(s, o); s2 += __shfl_down(s2, o); }
  __shared__ float red0[8], red1[8], mv[2];
  const int wid = threadIdx.x >> 5;
  if ((threadIdx.x & 31) == 0) { red0[wid] = s; red1[wid] = s2; }
  __syncthreads();
  if (threadIdx.x == 0) {
    float ts = 0.f, ts2 = 0.f;
    for (int i = 0; i < 8; ++i) { ts += red0[i]; ts2 += red1[i]; }
    float m = ts / N;
    mv[0] = m;
    mv[1] = rsqrtf(ts2 / N - m * m + 1e-5f);
  }
  __syncthreads();
  const float m = mv[0], inv = mv[1];
  for (int i = threadIdx.x; i < N; i += 256) {
    float v = (float)pa[i] + (pr ? al * (float)pr[i] : 0.0f);
    out[(size_t)row * N + i] = (bf16)((v - m) * inv * g[i] + beta[i]);
  }
}

// ----------------------------------------------------------------------------
// Fused attention, one wave per (b,h). Q/V/SW tiles via TDM (TENSORcnt) where
// available; K transposed via scatter staging. Softmax bias read from LDS.
// ----------------------------------------------------------------------------
__global__ __launch_bounds__(32)
void attention_kernel(const bf16* __restrict__ Qg, const bf16* __restrict__ Kg,
                      const bf16* __restrict__ Vg, const bf16* __restrict__ SWg,
                      const float* __restrict__ scale, bf16* __restrict__ Og) {
  __shared__ bf16 Qs[64][40];
  __shared__ bf16 Kt[32][72];
  __shared__ bf16 Vs[64][40];
  __shared__ bf16 Ps[64][72];
  __shared__ bf16 SWs[64][64];   // contiguous 8KB smolgen bias tile

  const int bh = blockIdx.x;
  const int b = bh / NH, h = bh % NH;
  const int lane = threadIdx.x;
  const int l16 = lane & 15, hk = lane >> 4;
  const size_t base = (size_t)b * NS * ND + (size_t)h * NDK;
  const bf16* qbase = Qg + base;
  const bf16* kbase = Kg + base;
  const bf16* vbase = Vg + base;
  const bf16* swp = SWg + (size_t)bh * (NS * NS);

#if HAVE_TDM
  // LDS byte offset = low 32 bits of the generic shared address (flat->LDS
  // mapping truncates, ISA 10.2). Pad fields reproduce the 80B row stride.
  tdm_load_2d((unsigned)(size_t)&Qs[0][0], qbase, NDK, NS, ND, 1);
  tdm_load_2d((unsigned)(size_t)&Vs[0][0], vbase, NDK, NS, ND, 1);
  tdm_load_2d((unsigned)(size_t)&SWs[0][0], swp, NS, NS, NS, 0);
#else
#pragma unroll
  for (int i = 0; i < 8; ++i) {
    int v = lane + i * 32;
    int row = v >> 2, seg = (v & 3) * 8;
    *(uint4*)&Qs[row][seg] = *(const uint4*)(qbase + (size_t)row * ND + seg);
    *(uint4*)&Vs[row][seg] = *(const uint4*)(vbase + (size_t)row * ND + seg);
  }
#pragma unroll
  for (int i = 0; i < 16; ++i) {   // SW tile: flat contiguous copy
    int v = lane + i * 32;
    ((uint4*)&SWs[0][0])[v] = ((const uint4*)swp)[v];
  }
#endif
#pragma unroll
  for (int i = 0; i < 8; ++i) {   // K transpose: Kt[d][key]
    int v = lane + i * 32;
    int row = v >> 2, seg = (v & 3) * 8;
    union { uint4 q; bf16 hv[8]; } u;
    u.q = *(const uint4*)(kbase + (size_t)row * ND + seg);
#pragma unroll
    for (int j = 0; j < 8; ++j) Kt[seg + j][row] = u.hv[j];
  }
#if HAVE_TDM
  __builtin_amdgcn_s_wait_tensorcnt(0);
#endif
  __syncthreads();

  floatx8 sacc[4][4] = {};
  {
    bhalf16 qf[4], kf[4];
#pragma unroll
    for (int i = 0; i < 4; ++i) qf[i] = load_a_frag(&Qs[i * 16 + l16][0], hk);
#pragma unroll
    for (int i = 0; i < 4; ++i) kf[i] = load_b_frag(&Kt[hk * 8][i * 16 + l16], 72);
#pragma unroll
    for (int fm = 0; fm < 4; ++fm)
#pragma unroll
      for (int fn = 0; fn < 4; ++fn)
        sacc[fm][fn] = wmma_bf16(qf[fm], kf[fn], sacc[fm][fn]);
  }

  const float sc = scale[0];

#pragma unroll
  for (int fm = 0; fm < 4; ++fm) {
#pragma unroll
    for (int r = 0; r < 8; ++r) {
      const int row = fm * 16 + r + hk * 8;
      float mx = -3.4e38f;
#pragma unroll
      for (int fn = 0; fn < 4; ++fn) {
        float v = sacc[fm][fn][r] * sc + (float)SWs[row][fn * 16 + l16];
        sacc[fm][fn][r] = v;
        mx = fmaxf(mx, v);
      }
#pragma unroll
      for (int o = 1; o < 16; o <<= 1) mx = fmaxf(mx, __shfl_xor(mx, o));
      float sum = 0.f;
#pragma unroll
      for (int fn = 0; fn < 4; ++fn) {
        float e = __expf(sacc[fm][fn][r] - mx);
        sacc[fm][fn][r] = e;
        sum += e;
      }
#pragma unroll
      for (int o = 1; o < 16; o <<= 1) sum += __shfl_xor(sum, o);
      float inv = 1.0f / sum;
#pragma unroll
      for (int fn = 0; fn < 4; ++fn)
        Ps[row][fn * 16 + l16] = (bf16)(sacc[fm][fn][r] * inv);
    }
  }
  __syncthreads();

  floatx8 oacc[4][2] = {};
#pragma unroll
  for (int ks = 0; ks < 2; ++ks) {
    bhalf16 pa[4], vb[2];
#pragma unroll
    for (int fm = 0; fm < 4; ++fm)
      pa[fm] = load_a_frag(&Ps[fm * 16 + l16][ks * 32], hk);
#pragma unroll
    for (int fn = 0; fn < 2; ++fn)
      vb[fn] = load_b_frag(&Vs[ks * 32 + hk * 8][fn * 16 + l16], 40);
#pragma unroll
    for (int fm = 0; fm < 4; ++fm)
#pragma unroll
      for (int fn = 0; fn < 2; ++fn)
        oacc[fm][fn] = wmma_bf16(pa[fm], vb[fn], oacc[fm][fn]);
  }

#pragma unroll
  for (int fm = 0; fm < 4; ++fm)
#pragma unroll
    for (int fn = 0; fn < 2; ++fn) {
      int col = fn * 16 + l16;
#pragma unroll
      for (int r = 0; r < 8; ++r) {
        int row = fm * 16 + r + hk * 8;
        Og[base + (size_t)row * ND + col] = (bf16)oacc[fm][fn][r];
      }
    }
}

// ----------------------------------------------------------------------------
// Policy scores, one wave per board: scores[b] = q[b] k[b]^T * scale (fp32 out)
// ----------------------------------------------------------------------------
__global__ __launch_bounds__(32)
void policy_scores_kernel(const bf16* __restrict__ Qg, const bf16* __restrict__ Kg,
                          const float* __restrict__ scale, float* __restrict__ Sg) {
  __shared__ bf16 Qs[64][40];
  __shared__ bf16 Kt[32][72];
  const int b = blockIdx.x;
  const int lane = threadIdx.x;
  const int l16 = lane & 15, hk = lane >> 4;

  floatx8 acc[4][4] = {};
  for (int k0 = 0; k0 < ND; k0 += 32) {
    const bf16* qb = Qg + (size_t)b * NS * ND + k0;
    const bf16* kb = Kg + (size_t)b * NS * ND + k0;
#pragma unroll
    for (int i = 0; i < 8; ++i) {
      int v = lane + i * 32;
      int row = v >> 2, seg = (v & 3) * 8;
      *(uint4*)&Qs[row][seg] = *(const uint4*)(qb + (size_t)row * ND + seg);
      union { uint4 q; bf16 hv[8]; } u;
      u.q = *(const uint4*)(kb + (size_t)row * ND + seg);
#pragma unroll
      for (int j = 0; j < 8; ++j) Kt[seg + j][row] = u.hv[j];
    }
    __syncthreads();
    bhalf16 qf[4], kf[4];
#pragma unroll
    for (int i = 0; i < 4; ++i) qf[i] = load_a_frag(&Qs[i * 16 + l16][0], hk);
#pragma unroll
    for (int i = 0; i < 4; ++i) kf[i] = load_b_frag(&Kt[hk * 8][i * 16 + l16], 72);
#pragma unroll
    for (int fm = 0; fm < 4; ++fm)
#pragma unroll
      for (int fn = 0; fn < 4; ++fn)
        acc[fm][fn] = wmma_bf16(qf[fm], kf[fn], acc[fm][fn]);
    __syncthreads();
  }

  const float sc = scale[0];
#pragma unroll
  for (int fm = 0; fm < 4; ++fm)
#pragma unroll
    for (int fn = 0; fn < 4; ++fn) {
      int col = fn * 16 + l16;
#pragma unroll
      for (int r = 0; r < 8; ++r) {
        int row = fm * 16 + r + hk * 8;
        Sg[(size_t)b * 4096 + row * 64 + col] = acc[fm][fn][r] * sc;
      }
    }
}

// promdot[b, c8, t] = k[b, 56+c8, :] . prom_w[t, :]
__global__ __launch_bounds__(32)
void promdot_kernel(const bf16* __restrict__ Kg, const float* __restrict__ promw,
                    float* __restrict__ pd) {
  const int b = blockIdx.x, t = threadIdx.x;
  const int c8 = t >> 2, j = t & 3;
  const bf16* kr = Kg + ((size_t)b * NS + 56 + c8) * ND;
  const float* wr = promw + (size_t)j * ND;
  float s = 0.f;
  for (int d = 0; d < ND; ++d) s = fmaf((float)kr[d], wr[d], s);
  pd[b * 32 + c8 * 4 + j] = s;
}

__global__ __launch_bounds__(256)
void policy_finalize_kernel(const float* __restrict__ scores,
                            const float* __restrict__ promdot,
                            const int* __restrict__ indices,
                            float* __restrict__ out) {
  const int gid = blockIdx.x * blockDim.x + threadIdx.x;
  if (gid >= NB * 1858) return;
  const int b = gid / 1858, i = gid % 1858;
  const int idx = indices[i];
  float val;
  if (idx < 4096) {
    val = scores[(size_t)b * 4096 + idx];
  } else {
    const int flat = idx - 4096;
    const int c24 = flat % 24;
    const int f64 = flat / 3;
    const int sr = 48 + (f64 >> 3), scol = 56 + (f64 & 7);
    const float sval = scores[(size_t)b * 4096 + sr * 64 + scol];
    const int c8 = c24 / 3, t = c24 % 3;
    const float pv = promdot[b * 32 + c8 * 4 + t] + promdot[b * 32 + c8 * 4 + 3];
    val = sval + pv;
  }
  out[gid] = val;
}

__global__ __launch_bounds__(256)
void softmax3_kernel(const float* __restrict__ in, float* __restrict__ out) {
  const int b = blockIdx.x * blockDim.x + threadIdx.x;
  if (b >= NB) return;
  float x0 = in[b * 3], x1 = in[b * 3 + 1], x2 = in[b * 3 + 2];
  float m = fmaxf(x0, fmaxf(x1, x2));
  float e0 = __expf(x0 - m), e1 = __expf(x1 - m), e2 = __expf(x2 - m);
  float inv = 1.0f / (e0 + e1 + e2);
  out[b * 3] = e0 * inv; out[b * 3 + 1] = e1 * inv; out[b * 3 + 2] = e2 * inv;
}

// ----------------------------------------------------------------------------
// Host orchestration.
// ----------------------------------------------------------------------------
static void launch_gemm_b(int act, const bf16* A, const bf16* Wt, const float* bias,
                          bf16* out, int M, int N, int Np, int K, hipStream_t s) {
  dim3 g((N + 127) / 128, (M + 127) / 128), b(128);
  switch (act) {
    case ACT_MISH:   gemm_v2<ACT_MISH,   bf16><<<g, b, 0, s>>>(A, Wt, bias, out, M, N, Np, K); break;
    case ACT_SILU:   gemm_v2<ACT_SILU,   bf16><<<g, b, 0, s>>>(A, Wt, bias, out, M, N, Np, K); break;
    case ACT_SQRELU: gemm_v2<ACT_SQRELU, bf16><<<g, b, 0, s>>>(A, Wt, bias, out, M, N, Np, K); break;
    default:         gemm_v2<ACT_NONE,   bf16><<<g, b, 0, s>>>(A, Wt, bias, out, M, N, Np, K); break;
  }
}
static void launch_gemm_f(int act, const bf16* A, const bf16* Wt, const float* bias,
                          float* out, int M, int N, int Np, int K, hipStream_t s) {
  dim3 g((N + 127) / 128, (M + 127) / 128), b(128);
  if (act == ACT_MISH)
    gemm_v2<ACT_MISH, float><<<g, b, 0, s>>>(A, Wt, bias, out, M, N, Np, K);
  else
    gemm_v2<ACT_NONE, float><<<g, b, 0, s>>>(A, Wt, bias, out, M, N, Np, K);
}

extern "C" void kernel_launch(void* const* d_in, const int* in_sizes, int n_in,
                              void* d_out, int out_size, void* d_ws, size_t ws_size,
                              hipStream_t stream) {
  const float* board   = (const float*)d_in[0];
  const float* emb_w   = (const float*)d_in[1];
  const float* emb_b   = (const float*)d_in[2];
  const float* gmul    = (const float*)d_in[3];
  const float* gadd    = (const float*)d_in[4];
  const float* pos     = (const float*)d_in[5];
  const float* l_qw    = (const float*)d_in[6];
  const float* l_qb    = (const float*)d_in[7];
  const float* l_kw    = (const float*)d_in[8];
  const float* l_kb    = (const float*)d_in[9];
  const float* l_vw    = (const float*)d_in[10];
  const float* l_vb    = (const float*)d_in[11];
  const float* l_ow    = (const float*)d_in[12];
  const float* l_ob    = (const float*)d_in[13];
  const float* l_qks   = (const float*)d_in[14];
  const float* l_sgcw  = (const float*)d_in[15];
  const float* l_sgd1w = (const float*)d_in[16];
  const float* l_sgd1b = (const float*)d_in[17];
  const float* l_sgl1g = (const float*)d_in[18];
  const float* l_sgl1b = (const float*)d_in[19];
  const float* l_sgd2w = (const float*)d_in[20];
  const float* l_sgd2b = (const float*)d_in[21];
  const float* l_sgl2g = (const float*)d_in[22];
  const float* l_sgl2b = (const float*)d_in[23];
  const float* l_sgwgw = (const float*)d_in[24];
  const float* l_ln1g  = (const float*)d_in[25];
  const float* l_ln1b  = (const float*)d_in[26];
  const float* l_ln2g  = (const float*)d_in[27];
  const float* l_ln2b  = (const float*)d_in[28];
  const float* l_f1w   = (const float*)d_in[29];
  const float* l_f1b   = (const float*)d_in[30];
  const float* l_f2w   = (const float*)d_in[31];
  const float* l_f2b   = (const float*)d_in[32];
  const float* l_ain   = (const float*)d_in[33];
  const float* l_aout  = (const float*)d_in[34];
  const float* p_d1w   = (const float*)d_in[35];
  const float* p_d1b   = (const float*)d_in[36];
  const float* p_qw    = (const float*)d_in[37];
  const float* p_qb    = (const float*)d_in[38];
  const float* p_kw    = (const float*)d_in[39];
  const float* p_kb    = (const float*)d_in[40];
  const float* p_scale = (const float*)d_in[41];
  const float* p_promw = (const float*)d_in[42];
  const float* v_ew    = (const float*)d_in[43];
  const float* v_eb    = (const float*)d_in[44];
  const float* v_d1w   = (const float*)d_in[45];
  const float* v_d1b   = (const float*)d_in[46];
  const float* v_d2w   = (const float*)d_in[47];
  const float* v_d2b   = (const float*)d_in[48];
  const float* m_ew    = (const float*)d_in[49];
  const float* m_eb    = (const float*)d_in[50];
  const float* m_d1w   = (const float*)d_in[51];
  const float* m_d1b   = (const float*)d_in[52];
  const float* m_d2w   = (const float*)d_in[53];
  const float* m_d2b   = (const float*)d_in[54];
  const int*   indices = (const int*)d_in[55];
  (void)in_sizes; (void)n_in; (void)out_size; (void)ws_size;

  char* wsb = (char*)d_ws;
  size_t off = 0;
  auto allocBytes = [&](size_t n) {
    void* p = wsb + off;
    off = (off + n + 255) & ~(size_t)255;
    return p;
  };
  auto aBF = [&](size_t e) { return (bf16*)allocBytes(e * 2); };
  auto aF  = [&](size_t e) { return (float*)allocBytes(e * 4); };

  // Activations (bf16).
  bf16* X   = aBF((size_t)NTOK * ND);
  bf16* X1  = aBF((size_t)NTOK * ND);
  bf16* Qb  = aBF((size_t)NTOK * ND);
  bf16* Kb  = aBF((size_t)NTOK * ND);
  bf16* Vb  = aBF((size_t)NTOK * ND);
  bf16* T0  = aBF((size_t)NTOK * NFF);
  bf16* T1  = aBF((size_t)NTOK * ND);
  bf16* SWb = aBF((size_t)NB * NH * NS * NS);
  bf16* SGC = aBF((size_t)NTOK * 32);
  bf16* SH1 = aBF((size_t)NB * 256);
  bf16* SH2 = aBF((size_t)NB * 6144);
  bf16* H1  = aBF((size_t)NB * 128);
  float* PS  = aF((size_t)NB * 4096);
  float* PD  = aF((size_t)NB * 32);
  float* WDL = aF((size_t)NB * 3);

  // Pre-transposed bf16 weights (K-major, N padded to >=128 multiples).
  bf16* WTq    = aBF((size_t)NL * ND * ND);
  bf16* WTk    = aBF((size_t)NL * ND * ND);
  bf16* WTv    = aBF((size_t)NL * ND * ND);
  bf16* WTo    = aBF((size_t)NL * ND * ND);
  bf16* WTsgc  = aBF((size_t)NL * ND * 128);
  bf16* WTsgd1 = aBF((size_t)NL * 2048 * 256);
  bf16* WTsgd2 = aBF((size_t)NL * 256 * 6144);
  bf16* WTsgwg = aBF((size_t)NL * 256 * 4096);
  bf16* WTf1   = aBF((size_t)NL * ND * NFF);
  bf16* WTf2   = aBF((size_t)NL * NFF * ND);
  bf16* WTpd1  = aBF((size_t)ND * ND);
  bf16* WTpq   = aBF((size_t)ND * ND);
  bf16* WTpk   = aBF((size_t)ND * ND);
  bf16* WTve   = aBF((size_t)ND * 128);
  bf16* WTvd1  = aBF((size_t)2048 * 128);
  bf16* WTvd2  = aBF((size_t)128 * 128);
  bf16* WTme   = aBF((size_t)ND * 128);
  bf16* WTmd1  = aBF((size_t)512 * 128);
  bf16* WTmd2  = aBF((size_t)128 * 128);

  auto tw = [&](const float* w, bf16* wt, int L, int N, int K, int Np) {
    dim3 g(Np / 32, K / 32, L), b(32, 8);
    transpose_tile_kernel<<<g, b, 0, stream>>>(w, wt, N, K, Np);
  };
  tw(l_qw,    WTq,    NL, ND,   ND,   ND);
  tw(l_kw,    WTk,    NL, ND,   ND,   ND);
  tw(l_vw,    WTv,    NL, ND,   ND,   ND);
  tw(l_ow,    WTo,    NL, ND,   ND,   ND);
  tw(l_sgcw,  WTsgc,  NL, 32,   ND,   128);
  tw(l_sgd1w, WTsgd1, NL, 256,  2048, 256);
  tw(l_sgd2w, WTsgd2, NL, 6144, 256,  6144);
  tw(l_sgwgw, WTsgwg, NL, 4096, 256,  4096);
  tw(l_f1w,   WTf1,   NL, NFF,  ND,   NFF);
  tw(l_f2w,   WTf2,   NL, ND,   NFF,  ND);
  tw(p_d1w,   WTpd1,  1,  ND,   ND,   ND);
  tw(p_qw,    WTpq,   1,  ND,   ND,   ND);
  tw(p_kw,    WTpk,   1,  ND,   ND,   ND);
  tw(v_ew,    WTve,   1,  32,   ND,   128);
  tw(v_d1w,   WTvd1,  1,  128,  2048, 128);
  tw(v_d2w,   WTvd2,  1,  3,    128,  128);
  tw(m_ew,    WTme,   1,  8,    ND,   128);
  tw(m_d1w,   WTmd1,  1,  128,  512,  128);
  tw(m_d2w,   WTmd2,  1,  1,    128,  128);

  embed_kernel<<<NTOK, 256, 0, stream>>>(board, pos, emb_w, emb_b, gmul, gadd, X);

  for (int l = 0; l < NL; ++l) {
    const bf16 *wtq = WTq + (size_t)l*ND*ND,   *wtk = WTk + (size_t)l*ND*ND;
    const bf16 *wtv = WTv + (size_t)l*ND*ND,   *wto = WTo + (size_t)l*ND*ND;
    const bf16 *wsgc = WTsgc + (size_t)l*ND*128;
    const bf16 *wsg1 = WTsgd1 + (size_t)l*2048*256;
    const bf16 *wsg2 = WTsgd2 + (size_t)l*256*6144;
    const bf16 *wswg = WTsgwg + (size_t)l*256*4096;
    const bf16 *wf1 = WTf1 + (size_t)l*ND*NFF, *wf2 = WTf2 + (size_t)l*NFF*ND;
    const float *qb = l_qb + (size_t)l*ND, *kb = l_kb + (size_t)l*ND;
    const float *vb = l_vb + (size_t)l*ND, *ob = l_ob + (size_t)l*ND;
    const float *qks = l_qks + l;
    const float *sg1b = l_sgd1b + (size_t)l*256;
    const float *sl1g = l_sgl1g + (size_t)l*256, *sl1b = l_sgl1b + (size_t)l*256;
    const float *sg2b = l_sgd2b + (size_t)l*6144;
    const float *sl2g = l_sgl2g + (size_t)l*6144, *sl2b = l_sgl2b + (size_t)l*6144;
    const float *n1g = l_ln1g + (size_t)l*ND, *n1b = l_ln1b + (size_t)l*ND;
    const float *n2g = l_ln2g + (size_t)l*ND, *n2b = l_ln2b + (size_t)l*ND;
    const float *f1b = l_f1b + (size_t)l*NFF, *f2b = l_f2b + (size_t)l*ND;
    const float *ain = l_ain + l, *aout = l_aout + l;

    // Smolgen.
    launch_gemm_b(ACT_NONE, X,   wsgc, nullptr, SGC, NTOK, 32,   128,  ND,   stream);
    launch_gemm_b(ACT_SILU, SGC, wsg1, sg1b,    SH1, NB,   256,  256,  2048, stream);
    ln_kernel<<<NB, 256, 0, stream>>>(SH1, nullptr, nullptr, sl1g, sl1b, SH1, 256);
    launch_gemm_b(ACT_SILU, SH1, wsg2, sg2b,    SH2, NB,   6144, 6144, 256,  stream);
    ln_kernel<<<NB, 256, 0, stream>>>(SH2, nullptr, nullptr, sl2g, sl2b, SH2, 6144);
    launch_gemm_b(ACT_NONE, SH2, wswg, nullptr, SWb, NB*NH, 4096, 4096, 256, stream);

    // Attention.
    launch_gemm_b(ACT_NONE, X, wtq, qb, Qb, NTOK, ND, ND, ND, stream);
    launch_gemm_b(ACT_NONE, X, wtk, kb, Kb, NTOK, ND, ND, ND, stream);
    launch_gemm_b(ACT_NONE, X, wtv, vb, Vb, NTOK, ND, ND, ND, stream);
    attention_kernel<<<NB * NH, 32, 0, stream>>>(Qb, Kb, Vb, SWb, qks, T0);
    launch_gemm_b(ACT_NONE, T0, wto, ob, T1, NTOK, ND, ND, ND, stream);
    ln_kernel<<<NTOK, 256, 0, stream>>>(T1, X, ain, n1g, n1b, X1, ND);

    // FFN.
    launch_gemm_b(ACT_SQRELU, X1, wf1, f1b, T0, NTOK, NFF, NFF, ND, stream);
    launch_gemm_b(ACT_NONE,   T0, wf2, f2b, T1, NTOK, ND,  ND,  NFF, stream);
    ln_kernel<<<NTOK, 256, 0, stream>>>(T1, X1, aout, n2g, n2b, X, ND);
  }

  float* out = (float*)d_out;

  // Policy head.
  launch_gemm_b(ACT_MISH, X,  WTpd1, p_d1b, T0, NTOK, ND, ND, ND, stream);
  launch_gemm_b(ACT_NONE, T0, WTpq,  p_qb,  Qb, NTOK, ND, ND, ND, stream);
  launch_gemm_b(ACT_NONE, T0, WTpk,  p_kb,  Kb, NTOK, ND, ND, ND, stream);
  policy_scores_kernel<<<NB, 32, 0, stream>>>(Qb, Kb, p_scale, PS);
  promdot_kernel<<<NB, 32, 0, stream>>>(Kb, p_promw, PD);
  policy_finalize_kernel<<<(NB * 1858 + 255) / 256, 256, 0, stream>>>(PS, PD, indices, out);

  // Value head.
  launch_gemm_b(ACT_MISH, X,   WTve,  v_eb,  SGC, NTOK, 32,  128, ND,   stream);
  launch_gemm_b(ACT_MISH, SGC, WTvd1, v_d1b, H1,  NB,   128, 128, 2048, stream);
  launch_gemm_f(ACT_NONE, H1,  WTvd2, v_d2b, WDL, NB,   3,   128, 128,  stream);
  softmax3_kernel<<<(NB + 255) / 256, 256, 0, stream>>>(WDL, out + (size_t)NB * 1858);

  // MLH head.
  launch_gemm_b(ACT_MISH, X,  WTme,  m_eb,  T0, NTOK, 8,   128, ND,  stream);
  launch_gemm_b(ACT_MISH, T0, WTmd1, m_d1b, H1, NB,   128, 128, 512, stream);
  launch_gemm_f(ACT_MISH, H1, WTmd2, m_d2b, out + (size_t)NB * 1858 + (size_t)NB * 3,
                NB, 1, 128, 128, stream);
}
